// Sender_LSTM_45329084842090
// MI455X (gfx1250) — matrix operand order repaired and checked by reading, not compiled, not run on hardware
//
#include <hip/hip_runtime.h>
#include <math.h>

// Problem constants (from reference): B=64, E=512, H=1024, V=32000, T=31
#define B_   64
#define E_   512
#define H_   1024
#define G4H  4096     // 4*H
#define V_   32000
#define T_   31
#define TT_  62       // 2*T (padded output width)

typedef __bf16 bf16_t;
typedef __attribute__((ext_vector_type(16))) __bf16 v16bf;
typedef __attribute__((ext_vector_type(8)))  __bf16 v8bf;
typedef __attribute__((ext_vector_type(8)))  float  v8f;

union BF16Frag { v16bf v; v8bf h[2]; };

// ---------------------------------------------------------------------------
// GEMM: C[64 x N] = A[64 x K](bf16, row-major) * BT[N x K](bf16, N-major)^T + bias
// 256 threads = 8 waves (4 M-tiles x 2 N-groups). Each wave: 16(M) x NT*16(N).
// Fragments double-buffered so global loads overlap WMMA issue.
// ---------------------------------------------------------------------------
template <int NT>
__global__ __launch_bounds__(256)
void gemm_bf16_wmma(const bf16_t* __restrict__ A, const bf16_t* __restrict__ BT,
                    const float* __restrict__ bias, float* __restrict__ C,
                    int K, int N)
{
    const int lane  = threadIdx.x & 31;
    const int wave  = threadIdx.x >> 5;
    const int half  = lane >> 4;     // lane group 0..15 vs 16..31
    const int l16   = lane & 15;
    const int mTile = (wave & 3) * 16;
    const int n0    = blockIdx.x * (NT * 32) + (wave >> 2) * (NT * 16);

    // A fragment (16x32 bf16): lane row m=l16; V0..3 hold K=half*16+0..7,
    // V4..7 hold K=half*16+8..15  -> two contiguous 16B loads.
    const bf16_t* aRow = A + (size_t)(mTile + l16) * K;
    // B fragment (32x16 bf16): lane col n=l16; lanes 0-15 K=0..15,
    // lanes 16-31 K=16..31 -> 16 contiguous K per lane (two 16B loads).
    const bf16_t* bRow[NT];
#pragma unroll
    for (int t = 0; t < NT; ++t)
        bRow[t] = BT + (size_t)(n0 + t * 16 + l16) * K + half * 16;

    v8f acc[NT];
#pragma unroll
    for (int t = 0; t < NT; ++t) acc[t] = (v8f){};

    BF16Frag a, b[NT], an, bn[NT];

    // prologue: load k0 = 0
    a.h[0] = *(const v8bf*)(aRow + half * 16);
    a.h[1] = *(const v8bf*)(aRow + half * 16 + 8);
#pragma unroll
    for (int t = 0; t < NT; ++t) {
        b[t].h[0] = *(const v8bf*)(bRow[t]);
        b[t].h[1] = *(const v8bf*)(bRow[t] + 8);
    }

    for (int k0 = 32; k0 < K; k0 += 32) {
        // issue next-iteration loads first (double buffer)
        an.h[0] = *(const v8bf*)(aRow + k0 + half * 16);
        an.h[1] = *(const v8bf*)(aRow + k0 + half * 16 + 8);
#pragma unroll
        for (int t = 0; t < NT; ++t) {
            bn[t].h[0] = *(const v8bf*)(bRow[t] + k0);
            bn[t].h[1] = *(const v8bf*)(bRow[t] + k0 + 8);
        }
        __builtin_prefetch(bRow[0] + k0 + 32, 0, 3);          // stream weights ahead
        __builtin_prefetch(bRow[NT - 1] + k0 + 32, 0, 3);
#pragma unroll
        for (int t = 0; t < NT; ++t)
            acc[t] = __builtin_amdgcn_wmma_f32_16x16x32_bf16(false, a.v, false, b[t].v,
                                                             (short)0, acc[t], false, false);
        a = an;
#pragma unroll
        for (int t = 0; t < NT; ++t) b[t] = bn[t];
    }
#pragma unroll
    for (int t = 0; t < NT; ++t)
        acc[t] = __builtin_amdgcn_wmma_f32_16x16x32_bf16(false, a.v, false, b[t].v,
                                                         (short)0, acc[t], false, false);

    // C/D layout: VGPR r -> M = r + 8*half ; N = lane&15
#pragma unroll
    for (int t = 0; t < NT; ++t) {
        const int   n  = n0 + t * 16 + l16;
        const float bi = bias[n];
#pragma unroll
        for (int r = 0; r < 8; ++r) {
            const int m = mTile + r + 8 * half;
            C[(size_t)m * N + n] = acc[t][r] + bi;
        }
    }
}

// ---------------------------------------------------------------------------
// Transpose + (optional sum of two inputs) + fp32 -> bf16.  in: R x Cc row-major
// out: Cc x R (bf16).  blockDim (32,8), LDS tile padded to kill bank conflicts.
// ---------------------------------------------------------------------------
__global__ __launch_bounds__(256)
void transpose_cvt_bf16(const float* __restrict__ in, const float* __restrict__ in2,
                        bf16_t* __restrict__ out, int R, int Cc, int addTwo)
{
    __shared__ float tile[32][33];
    const int c0 = blockIdx.x * 32;
    const int r0 = blockIdx.y * 32;
    for (int i = threadIdx.y; i < 32; i += 8) {
        size_t idx = (size_t)(r0 + i) * Cc + c0 + threadIdx.x;
        float v = in[idx];
        if (addTwo) v += in2[idx];
        tile[i][threadIdx.x] = v;
    }
    __syncthreads();
    for (int i = threadIdx.y; i < 32; i += 8)
        out[(size_t)(c0 + i) * R + (r0 + threadIdx.x)] = (bf16_t)tile[threadIdx.x][i];
}

__global__ void cvt_f32_bf16(const float* __restrict__ in, bf16_t* __restrict__ out, int n)
{
    int i = blockIdx.x * blockDim.x + threadIdx.x;
    if (i < n) out[i] = (bf16_t)in[i];
}

__global__ void zero_f32(float* __restrict__ p, int n)
{
    int i = blockIdx.x * blockDim.x + threadIdx.x;
    if (i < n) p[i] = 0.0f;
}

__global__ void copy_f32(const float* __restrict__ in, float* __restrict__ out, int n)
{
    int i = blockIdx.x * blockDim.x + threadIdx.x;
    if (i < n) out[i] = in[i];
}

// ---------------------------------------------------------------------------
// LSTM pointwise: gate order i,f,g,o ; g and cell-output linear, gates sigmoid.
// ---------------------------------------------------------------------------
__global__ __launch_bounds__(256)
void lstm_pointwise(const float* __restrict__ z, float* __restrict__ c,
                    float* __restrict__ h, bf16_t* __restrict__ h_bf, int useC)
{
    const int idx = blockIdx.x * blockDim.x + threadIdx.x;   // < 64*1024
    const int b = idx >> 10, j = idx & 1023;
    const float* zr = z + (size_t)b * G4H;
    const float zi = zr[j], zf = zr[H_ + j], zg = zr[2 * H_ + j], zo = zr[3 * H_ + j];
    const float cp = useC ? c[idx] : 0.0f;
    const float si = 1.0f / (1.0f + expf(-zi));
    const float sf = 1.0f / (1.0f + expf(-zf));
    const float so = 1.0f / (1.0f + expf(-zo));
    const float cn = sf * cp + si * zg;
    const float hn = so * cn;
    c[idx] = cn; h[idx] = hn; h_bf[idx] = (bf16_t)hn;
}

// ---------------------------------------------------------------------------
// Per-row log-softmax stats + Gumbel-max categorical sample + entropy.
// One block per batch row; two passes over V=32000.
// ---------------------------------------------------------------------------
__device__ __forceinline__ unsigned mix32(unsigned x)
{
    x ^= x >> 17; x *= 0xed5ad4bbu;
    x ^= x >> 11; x *= 0xac4c1b51u;
    x ^= x >> 15; x *= 0x31848babu;
    x ^= x >> 14;
    return x;
}

__global__ __launch_bounds__(256)
void row_softmax_sample(const float* __restrict__ logits,
                        float* __restrict__ outMsg, float* __restrict__ outLp,
                        float* __restrict__ outEnt, int t)
{
    __shared__ float sMax[256];
    __shared__ float sBest[256];
    __shared__ int   sIdx[256];
    __shared__ float sSum[256];
    __shared__ float sW[256];

    const int b = blockIdx.x, tid = threadIdx.x;
    const float* row = logits + (size_t)b * V_;

    float mx = -3.4e38f, best = -3.4e38f; int bi = 0;
    for (int v = tid; v < V_; v += 256) {
        const float l = row[v];
        mx = fmaxf(mx, l);
        unsigned x = mix32(((unsigned)t * 1315423911u) ^
                           ((unsigned)b * 2654435761u) ^
                           ((unsigned)v * 0x9E3779B9u) ^ 0x1234u);
        float u = fmaxf((float)(x >> 8) * (1.0f / 16777216.0f), 1e-10f);
        const float g = -logf(-logf(u) + 1e-20f);      // Gumbel(0,1)
        const float p = l + g;
        if (p > best) { best = p; bi = v; }
    }
    sMax[tid] = mx; sBest[tid] = best; sIdx[tid] = bi;
    __syncthreads();
    for (int s = 128; s > 0; s >>= 1) {
        if (tid < s) {
            sMax[tid] = fmaxf(sMax[tid], sMax[tid + s]);
            if (sBest[tid + s] > sBest[tid]) { sBest[tid] = sBest[tid + s]; sIdx[tid] = sIdx[tid + s]; }
        }
        __syncthreads();
    }
    const float rowMax = sMax[0];
    const int   sample = sIdx[0];

    float sum = 0.0f, w = 0.0f;
    for (int v = tid; v < V_; v += 256) {
        const float d = row[v] - rowMax;
        const float e = expf(d);
        sum += e; w += d * e;
    }
    sSum[tid] = sum; sW[tid] = w;
    __syncthreads();
    for (int s = 128; s > 0; s >>= 1) {
        if (tid < s) { sSum[tid] += sSum[tid + s]; sW[tid] += sW[tid + s]; }
        __syncthreads();
    }
    if (tid == 0) {
        const float Z = sSum[0];
        const float logZ = logf(Z);
        outEnt[(size_t)b * TT_ + t] = logZ - sW[0] / Z;                 // H = logZ - E[l-m]
        outLp [(size_t)b * TT_ + t] = row[sample] - rowMax - logZ;      // log p(sample)
        outMsg[(size_t)b * TT_ + t] = (float)sample;
    }
}

// ---------------------------------------------------------------------------
extern "C" void kernel_launch(void* const* d_in, const int* in_sizes, int n_in,
                              void* d_out, int out_size, void* d_ws, size_t ws_size,
                              hipStream_t stream)
{
    const float* inp = (const float*)d_in[0];   // (64,1,512)
    const float* Wx1 = (const float*)d_in[1];   // (512,4096)
    // d_in[2] = Wh1 : unused (encoder h0 == 0, single step)
    const float* b1  = (const float*)d_in[3];   // (4096)
    const float* Wx2 = (const float*)d_in[4];   // (1024,4096)
    const float* Wh2 = (const float*)d_in[5];   // (1024,4096)
    const float* b2  = (const float*)d_in[6];   // (4096)
    const float* Wd  = (const float*)d_in[7];   // (1024,32000)
    const float* bd  = (const float*)d_in[8];   // (32000)
    float* out = (float*)d_out;

    // Workspace carve-out
    char* ws = (char*)d_ws;
    size_t off = 0;
    auto carve = [&](size_t bytes) -> void* {
        void* p = ws + off;
        off = (off + bytes + 255) & ~(size_t)255;
        return p;
    };
    bf16_t* wx1t   = (bf16_t*)carve((size_t)G4H * E_ * 2);   //  4 MB  (4096 x 512)
    bf16_t* w2t    = (bf16_t*)carve((size_t)G4H * H_ * 2);   //  8 MB  (4096 x 1024)
    bf16_t* wdt    = (bf16_t*)carve((size_t)V_  * H_ * 2);   // 64 MB  (32000 x 1024) -> lives in L2
    bf16_t* x_bf   = (bf16_t*)carve((size_t)B_  * E_ * 2);
    bf16_t* h_bf   = (bf16_t*)carve((size_t)B_  * H_ * 2);
    float*  z      = (float*) carve((size_t)B_  * G4H * 4);
    float*  c      = (float*) carve((size_t)B_  * H_ * 4);
    float*  h      = (float*) carve((size_t)B_  * H_ * 4);
    float*  logits = (float*) carve((size_t)B_  * V_ * 4);   //  8 MB

    const dim3 tb(32, 8);
    // One-time weight prep: bf16 + N-major transpose (Wx2+Wh2 folded)
    transpose_cvt_bf16<<<dim3(G4H / 32, E_ / 32), tb, 0, stream>>>(Wx1, Wx1, wx1t, E_, G4H, 0);
    transpose_cvt_bf16<<<dim3(G4H / 32, H_ / 32), tb, 0, stream>>>(Wx2, Wh2, w2t, H_, G4H, 1);
    transpose_cvt_bf16<<<dim3(V_  / 32, H_ / 32), tb, 0, stream>>>(Wd, Wd, wdt, H_, V_, 0);
    cvt_f32_bf16<<<(B_ * E_) / 256, 256, 0, stream>>>(inp, x_bf, B_ * E_);
    zero_f32<<<(3 * B_ * TT_ + 255) / 256, 256, 0, stream>>>(out, 3 * B_ * TT_);

    // Encoder: single step with h0 = c0 = 0  =>  z = x@Wx1 + b1
    gemm_bf16_wmma<2><<<G4H / 64, 256, 0, stream>>>(x_bf, wx1t, b1, z, E_, G4H);
    lstm_pointwise<<<(B_ * H_) / 256, 256, 0, stream>>>(z, c, h, h_bf, 0);

    // Decoder: 31 sequential steps
    for (int t = 0; t < T_; ++t) {
        // gates: z = h @ (Wx2+Wh2) + b2   (64x1024)x(1024x4096)
        gemm_bf16_wmma<2><<<G4H / 64, 256, 0, stream>>>(h_bf, w2t, b2, z, H_, G4H);
        lstm_pointwise<<<(B_ * H_) / 256, 256, 0, stream>>>(z, c, h, h_bf, 1);
        // logits: (64x1024)x(1024x32000), wide tile to cut A re-reads
        gemm_bf16_wmma<4><<<V_ / 128, 256, 0, stream>>>(h_bf, wdt, bd, logits, H_, V_);
        row_softmax_sample<<<B_, 256, 0, stream>>>(logits, out, out + B_ * TT_,
                                                   out + 2 * B_ * TT_, t);
    }

    // Final hidden state -> 4th output
    copy_f32<<<(B_ * H_) / 256, 256, 0, stream>>>(h, out + 3 * B_ * TT_, B_ * H_);
}
